// CHGNetGraphConv_66649302499833
// MI455X (gfx1250) — matrix operand order repaired and checked by reading, not compiled, not run on hardware
//
#include <hip/hip_runtime.h>
#include <stdint.h>

// ---------------------------------------------------------------------------
// CHGNet graph-conv fused kernel for MI455X (gfx1250, wave32, WMMA bf16).
// ---------------------------------------------------------------------------

typedef __attribute__((ext_vector_type(16))) __bf16 v16bf;
typedef __attribute__((ext_vector_type(8)))  float  v8f;
typedef uint32_t __attribute__((may_alias)) u32a;

static constexpr int kN = 50000;
static constexpr int kE = 500000;
static constexpr int kD = 64;
static constexpr int kR = 9;

static constexpr int INP_STRIDE  = 200;  // bf16 elems/row for [16][192] tile (padded)
static constexpr int ACT_STRIDE  = 72;   // bf16 elems/row for [16][64] tile (padded)
static constexpr int RBF_STRIDE  = 40;   // bf16 elems/row for [16][32] zero-padded rbf tile
static constexpr int WAVES       = 8;    // waves per workgroup (256 threads)

// Fragment regions (units of fragments; 1 fragment = 32 lanes x 16 bf16 = 512)
static constexpr int FB_EW1  = 0;    // e_W1  [192x64] : 6 ksteps * 4 ntiles = 24
static constexpr int FB_EW2  = 24;   // e_W2  [ 64x64] : 2*4 = 8
static constexpr int FB_EGW1 = 32;   // e_gW1 : 24
static constexpr int FB_EGW2 = 56;   // e_gW2 : 8
static constexpr int FB_NW1  = 64;   // n_W1  : 24
static constexpr int FB_NW2  = 88;   // n_W2  : 8
static constexpr int FB_NGW1 = 96;   // n_gW1 : 24
static constexpr int FB_NGW2 = 120;  // n_gW2 : 8
static constexpr int FB_OUTW = 128;  // node_out_W : 8
static constexpr int FB_EWF  = 136;  // edge_wf_W [9x64] zero-padded to K=32 : 4
static constexpr int FB_NWF  = 140;  // node_wf_W [9x64] zero-padded to K=32 : 4
static constexpr int N_FRAGS = 144;

__device__ __forceinline__ float sigmoidf_(float x) { return 1.0f / (1.0f + __expf(-x)); }
__device__ __forceinline__ float siluf_(float x)    { return x * sigmoidf_(x); }

__device__ __forceinline__ uint32_t pk2bf(float a, float b) {
  union { uint32_t u; __bf16 h[2]; } x;
  x.h[0] = (__bf16)a; x.h[1] = (__bf16)b;
  return x.u;
}

union AFrag { v16bf v; u32a u[8]; };

// A fragment (16x32 bf16) gathered from an LDS row-major [16][stride] tile.
// Contiguous k order (B uses the same): lanes 0-15 -> M=lane, K = 0..15; lanes 16-31 -> K = 16..31.
__device__ __forceinline__ v16bf load_a_frag(const __bf16* A, int stride, int lane, int kbase) {
  const int m = lane & 15, khalf = lane >> 4;
  const __bf16* row = A + m * stride + kbase + khalf * 16;
  AFrag a;
#pragma unroll
  for (int p = 0; p < 4; ++p) a.u[p]     = *reinterpret_cast<const u32a*>(row + 2 * p);
#pragma unroll
  for (int p = 0; p < 4; ++p) a.u[4 + p] = *reinterpret_cast<const u32a*>(row + 8 + 2 * p);
  return a.v;
}

// [16 x K] (LDS bf16) @ [K x 64] (pre-laid-out global bf16 fragments) -> 4x v8f acc.
template <int KSTEPS>
__device__ __forceinline__ void mm_tile(const __bf16* A, int stride,
                                        const __bf16* frag, int base, int lane,
                                        v8f acc[4]) {
#pragma unroll
  for (int ks = 0; ks < KSTEPS; ++ks) {
    const v16bf a = load_a_frag(A, stride, lane, ks * 32);
#pragma unroll
    for (int nt = 0; nt < 4; ++nt) {
      const v16bf b = *reinterpret_cast<const v16bf*>(
          frag + (size_t)(base + nt * KSTEPS + ks) * 512 + lane * 16);
      acc[nt] = __builtin_amdgcn_wmma_f32_16x16x32_bf16(
          /*neg_a=*/false, a, /*neg_b=*/false, b,
          /*c_mod=*/(short)0, acc[nt], /*reuse_a=*/false, /*reuse_b=*/false);
    }
  }
}

// C layout: VGPR r, lanes 0-15 -> M=r, lanes 16-31 -> M=r+8; N = lane%16 (+16*ntile).
__device__ __forceinline__ void store_act_silu(__bf16* act, int lane, const v8f acc[4],
                                               const float* bias) {
  const int n15 = lane & 15, hf = lane >> 4;
#pragma unroll
  for (int nt = 0; nt < 4; ++nt) {
    const int col = n15 + nt * 16;
    const float b = bias[col];
#pragma unroll
    for (int r = 0; r < 8; ++r)
      act[(r + 8 * hf) * ACT_STRIDE + col] = (__bf16)siluf_(acc[nt][r] + b);
  }
}

// out = silu(silu(x@W1+b1)@W2+b2) * sigmoid(silu(x@G1+gb1)@G2+gb2)   (16 rows, K=192)
__device__ __forceinline__ void gated_mlp_tile(const __bf16* inp, __bf16* act,
                                               const __bf16* frag,
                                               int bW1, int bW2, int bG1, int bG2,
                                               const float* b1, const float* b2,
                                               const float* gb1, const float* gb2,
                                               int lane, float hg[4][8]) {
  const v8f vz = {0.f, 0.f, 0.f, 0.f, 0.f, 0.f, 0.f, 0.f};
  const int n15 = lane & 15;
  v8f acc[4] = {vz, vz, vz, vz};
  mm_tile<6>(inp, INP_STRIDE, frag, bW1, lane, acc);
  store_act_silu(act, lane, acc, b1);
  acc[0] = vz; acc[1] = vz; acc[2] = vz; acc[3] = vz;
  mm_tile<2>(act, ACT_STRIDE, frag, bW2, lane, acc);
  float h[4][8];
#pragma unroll
  for (int nt = 0; nt < 4; ++nt) {
    const float b = b2[n15 + nt * 16];
#pragma unroll
    for (int r = 0; r < 8; ++r) h[nt][r] = siluf_(acc[nt][r] + b);
  }
  acc[0] = vz; acc[1] = vz; acc[2] = vz; acc[3] = vz;
  mm_tile<6>(inp, INP_STRIDE, frag, bG1, lane, acc);
  store_act_silu(act, lane, acc, gb1);
  acc[0] = vz; acc[1] = vz; acc[2] = vz; acc[3] = vz;
  mm_tile<2>(act, ACT_STRIDE, frag, bG2, lane, acc);
#pragma unroll
  for (int nt = 0; nt < 4; ++nt) {
    const float b = gb2[n15 + nt * 16];
#pragma unroll
    for (int r = 0; r < 8; ++r) hg[nt][r] = h[nt][r] * sigmoidf_(acc[nt][r] + b);
  }
}

// ---------------------------------------------------------------------------
// Weight prep: f32 [K x 64] -> bf16 B-fragments. One 32-lane block per fragment.
// B layout (32x16 per fragment): lane = n + 16*((k%32)/16), element j = k%16.
// wf matrices are zero-padded from K=9 to K=32.
// ---------------------------------------------------------------------------
__global__ void chgnet_prep_weights(const float* eW1, const float* eW2,
                                    const float* egW1, const float* egW2,
                                    const float* nW1, const float* nW2,
                                    const float* ngW1, const float* ngW2,
                                    const float* outW, const float* ewf,
                                    const float* nwf, __bf16* frag) {
  const int f = blockIdx.x;
  const int lane = threadIdx.x;
  const float* W; int K; int base; int kvalid;
  if      (f < FB_EW2)  { W = eW1;  K = 192; base = FB_EW1;  kvalid = 192; }
  else if (f < FB_EGW1) { W = eW2;  K = 64;  base = FB_EW2;  kvalid = 64;  }
  else if (f < FB_EGW2) { W = egW1; K = 192; base = FB_EGW1; kvalid = 192; }
  else if (f < FB_NW1)  { W = egW2; K = 64;  base = FB_EGW2; kvalid = 64;  }
  else if (f < FB_NW2)  { W = nW1;  K = 192; base = FB_NW1;  kvalid = 192; }
  else if (f < FB_NGW1) { W = nW2;  K = 64;  base = FB_NW2;  kvalid = 64;  }
  else if (f < FB_NGW2) { W = ngW1; K = 192; base = FB_NGW1; kvalid = 192; }
  else if (f < FB_OUTW) { W = ngW2; K = 64;  base = FB_NGW2; kvalid = 64;  }
  else if (f < FB_EWF)  { W = outW; K = 64;  base = FB_OUTW; kvalid = 64;  }
  else if (f < FB_NWF)  { W = ewf;  K = 32;  base = FB_EWF;  kvalid = kR;  }
  else                  { W = nwf;  K = 32;  base = FB_NWF;  kvalid = kR;  }
  const int lf = f - base;
  const int ksteps = K / 32;
  const int ntile = lf / ksteps;
  const int kstep = lf % ksteps;
  const int n = (lane & 15) + ntile * 16;
  const int khalf = lane >> 4;
  __bf16* dst = frag + (size_t)f * 512 + lane * 16;
#pragma unroll
  for (int j = 0; j < 16; ++j) {
    const int k = kstep * 32 + khalf * 16 + j;
    dst[j] = (k < kvalid) ? (__bf16)W[k * kD + n] : (__bf16)0.0f;
  }
}

__global__ void chgnet_zero(float4* p, int n4) {
  const int i = blockIdx.x * blockDim.x + threadIdx.x;
  if (i < n4) p[i] = make_float4(0.f, 0.f, 0.f, 0.f);
}

// ---------------------------------------------------------------------------
// Fused edge + message kernel. One wave per 16-edge tile.
// ---------------------------------------------------------------------------
struct EdgeParams {
  const float* node_f; const float* edge_f;
  const int* src; const int* dst;
  const float* rbf; const float* snw; const float* sew;
  const float* e_b1; const float* e_b2; const float* e_gb1; const float* e_gb2;
  const float* n_b1; const float* n_b2; const float* n_gb1; const float* n_gb2;
  const __bf16* frag;
  float* agg; float* out_edge;
};

__global__ __launch_bounds__(WAVES * 32) void chgnet_edge_kernel(EdgeParams P) {
  __shared__ __bf16 inp_s[WAVES][16 * INP_STRIDE];
  __shared__ __bf16 act_s[WAVES][16 * ACT_STRIDE];
  __shared__ __bf16 rbf_s[WAVES][16 * RBF_STRIDE];

  const int wave = threadIdx.x >> 5;
  const int lane = threadIdx.x & 31;
  const int tile = blockIdx.x * WAVES + wave;
  if (tile >= kE / 16) return;
  const int e0 = tile * 16;

  __bf16* inp = inp_s[wave];
  __bf16* act = act_s[wave];
  __bf16* rbw = rbf_s[wave];

  const int m  = lane & 15;
  const int hf = lane >> 4;

  // ---- stage [xi | edge | xj] as bf16 into LDS; rbf as zero-padded bf16 tile ----
  {
    const int sI = P.src[e0 + m];
    const int dI = P.dst[e0 + m];
    const float* xi = P.node_f + (size_t)sI * kD + hf * 32;
    const float* xj = P.node_f + (size_t)dI * kD + hf * 32;
    const float* ef = P.edge_f + (size_t)(e0 + m) * kD + hf * 32;
    __bf16* row = inp + m * INP_STRIDE + hf * 32;
#pragma unroll
    for (int c = 0; c < 32; c += 2) {
      *reinterpret_cast<u32a*>(row + c)       = pk2bf(xi[c], xi[c + 1]);
      *reinterpret_cast<u32a*>(row + 64 + c)  = pk2bf(ef[c], ef[c + 1]);
      *reinterpret_cast<u32a*>(row + 128 + c) = pk2bf(xj[c], xj[c + 1]);
    }
    // rbf tile: zero K=0..31 then overwrite K=0..8 (in-order LDS within wave)
    __bf16* zr = rbw + m * RBF_STRIDE + hf * 16;
#pragma unroll
    for (int c = 0; c < 16; c += 2) *reinterpret_cast<u32a*>(zr + c) = 0u;
    if (hf == 0) {
      const float* rp = P.rbf + (size_t)(e0 + m) * kR;
      __bf16* rw = rbw + m * RBF_STRIDE;
#pragma unroll
      for (int q = 0; q < 8; q += 2)
        *reinterpret_cast<u32a*>(rw + q) = pk2bf(rp[q], rp[q + 1]);
      *reinterpret_cast<u32a*>(rw + 8) = pk2bf(rp[8], 0.f);
    }
    // prefetch the late-epilogue streams while WMMAs run
    __builtin_prefetch(P.sew + (size_t)(e0 + m) * kD + hf * 32, 0, 3);
    __builtin_prefetch(P.snw + (size_t)(e0 + m) * kD + hf * 32, 0, 3);
  }

  const int n15 = lane & 15;
  const v8f vz = {0.f, 0.f, 0.f, 0.f, 0.f, 0.f, 0.f, 0.f};
  float hg[4][8];

  // ---- edge (bond) gated MLP ----
  gated_mlp_tile(inp, act, P.frag, FB_EW1, FB_EW2, FB_EGW1, FB_EGW2,
                 P.e_b1, P.e_b2, P.e_gb1, P.e_gb2, lane, hg);

  // ---- rbf @ edge_wf_W via WMMA (K padded 9->32) ----
  v8f wacc[4] = {vz, vz, vz, vz};
  mm_tile<1>(rbw, RBF_STRIDE, P.frag, FB_EWF, lane, wacc);

  // ---- edge epilogue: * (rbf@edge_wf) * sew, residual, store; refresh LDS mid ----
#pragma unroll
  for (int r = 0; r < 8; ++r) {
    const int row = r + 8 * hf;
    const size_t e = (size_t)(e0 + row);
#pragma unroll
    for (int nt = 0; nt < 4; ++nt) {
      const int col = n15 + nt * 16;
      const float eu = hg[nt][r] * wacc[nt][r] * P.sew[e * kD + col];
      const float ne = P.edge_f[e * kD + col] + eu;
      P.out_edge[e * kD + col] = ne;
      inp[row * INP_STRIDE + 64 + col] = (__bf16)ne;  // new_edge for node MLP input
    }
  }

  // ---- node (atom) gated MLP on [xi | new_edge | xj] ----
  gated_mlp_tile(inp, act, P.frag, FB_NW1, FB_NW2, FB_NGW1, FB_NGW2,
                 P.n_b1, P.n_b2, P.n_gb1, P.n_gb2, lane, hg);

  // ---- rbf @ node_wf_W via WMMA ----
  wacc[0] = vz; wacc[1] = vz; wacc[2] = vz; wacc[3] = vz;
  mm_tile<1>(rbw, RBF_STRIDE, P.frag, FB_NWF, lane, wacc);

  // ---- message epilogue: * (rbf@node_wf) * snw, scatter-add at dst ----
#pragma unroll
  for (int r = 0; r < 8; ++r) {
    const int row = r + 8 * hf;
    const size_t e = (size_t)(e0 + row);
    const int d2 = P.dst[e0 + row];
#pragma unroll
    for (int nt = 0; nt < 4; ++nt) {
      const int col = n15 + nt * 16;
      const float msg = hg[nt][r] * wacc[nt][r] * P.snw[e * kD + col];
      __hip_atomic_fetch_add(&P.agg[(size_t)d2 * kD + col], msg,
                             __ATOMIC_RELAXED, __HIP_MEMORY_SCOPE_AGENT);
    }
  }
}

// ---------------------------------------------------------------------------
// Node output: new_node = node + agg @ node_out_W   (WMMA, one wave per 16 rows)
// ---------------------------------------------------------------------------
__global__ __launch_bounds__(WAVES * 32) void chgnet_node_kernel(
    const float* node_f, const float* agg, const __bf16* frag, float* out_node) {
  __shared__ __bf16 a_s[WAVES][16 * ACT_STRIDE];
  const int wave = threadIdx.x >> 5;
  const int lane = threadIdx.x & 31;
  const int tile = blockIdx.x * WAVES + wave;
  if (tile >= kN / 16) return;
  const int r0 = tile * 16;
  __bf16* buf = a_s[wave];
  const int m = lane & 15, hf = lane >> 4;
  {
    const float* ar = agg + (size_t)(r0 + m) * kD + hf * 32;
    __bf16* row = buf + m * ACT_STRIDE + hf * 32;
#pragma unroll
    for (int c = 0; c < 32; c += 2)
      *reinterpret_cast<u32a*>(row + c) = pk2bf(ar[c], ar[c + 1]);
  }
  const v8f vz = {0.f, 0.f, 0.f, 0.f, 0.f, 0.f, 0.f, 0.f};
  v8f acc[4] = {vz, vz, vz, vz};
  mm_tile<2>(buf, ACT_STRIDE, frag, FB_OUTW, lane, acc);
  const int n15 = lane & 15;
#pragma unroll
  for (int nt = 0; nt < 4; ++nt) {
#pragma unroll
    for (int r = 0; r < 8; ++r) {
      const size_t idx = (size_t)(r0 + r + 8 * hf) * kD + (n15 + nt * 16);
      out_node[idx] = node_f[idx] + acc[nt][r];
    }
  }
}

// ---------------------------------------------------------------------------
extern "C" void kernel_launch(void* const* d_in, const int* in_sizes, int n_in,
                              void* d_out, int out_size, void* d_ws, size_t ws_size,
                              hipStream_t stream) {
  (void)in_sizes; (void)n_in; (void)out_size; (void)ws_size;

  const float* node_f = (const float*)d_in[0];
  const float* edge_f = (const float*)d_in[1];
  const int*   src    = (const int*)d_in[2];
  const int*   dst    = (const int*)d_in[3];
  const float* rbf    = (const float*)d_in[4];
  const float* snw    = (const float*)d_in[5];
  const float* sew    = (const float*)d_in[6];
  const float* eW1  = (const float*)d_in[7];
  const float* eb1  = (const float*)d_in[8];
  const float* eW2  = (const float*)d_in[9];
  const float* eb2  = (const float*)d_in[10];
  const float* egW1 = (const float*)d_in[11];
  const float* egb1 = (const float*)d_in[12];
  const float* egW2 = (const float*)d_in[13];
  const float* egb2 = (const float*)d_in[14];
  const float* nW1  = (const float*)d_in[15];
  const float* nb1  = (const float*)d_in[16];
  const float* nW2  = (const float*)d_in[17];
  const float* nb2  = (const float*)d_in[18];
  const float* ngW1 = (const float*)d_in[19];
  const float* ngb1 = (const float*)d_in[20];
  const float* ngW2 = (const float*)d_in[21];
  const float* ngb2 = (const float*)d_in[22];
  const float* outW = (const float*)d_in[23];
  const float* nwf  = (const float*)d_in[24];
  const float* ewf  = (const float*)d_in[25];

  float* out_node = (float*)d_out;
  float* out_edge = (float*)d_out + (size_t)kN * kD;

  // Workspace: [0, N*64*4) = agg f32; then bf16 weight fragments (32B aligned).
  float*  agg  = (float*)d_ws;
  __bf16* frag = (__bf16*)((char*)d_ws + (size_t)kN * kD * sizeof(float));

  chgnet_prep_weights<<<N_FRAGS, 32, 0, stream>>>(eW1, eW2, egW1, egW2,
                                                  nW1, nW2, ngW1, ngW2, outW,
                                                  ewf, nwf, frag);
  chgnet_zero<<<(kN * kD / 4 + 255) / 256, 256, 0, stream>>>((float4*)agg, kN * kD / 4);

  EdgeParams P;
  P.node_f = node_f; P.edge_f = edge_f; P.src = src; P.dst = dst;
  P.rbf = rbf; P.snw = snw; P.sew = sew;
  P.e_b1 = eb1; P.e_b2 = eb2; P.e_gb1 = egb1; P.e_gb2 = egb2;
  P.n_b1 = nb1; P.n_b2 = nb2; P.n_gb1 = ngb1; P.n_gb2 = ngb2;
  P.frag = frag; P.agg = agg; P.out_edge = out_edge;

  const int etiles = kE / 16;  // 31250
  chgnet_edge_kernel<<<(etiles + WAVES - 1) / WAVES, WAVES * 32, 0, stream>>>(P);

  const int ntiles = kN / 16;  // 3125
  chgnet_node_kernel<<<(ntiles + WAVES - 1) / WAVES, WAVES * 32, 0, stream>>>(
      node_f, agg, frag, out_node);
}
